// MultimodalPDE_14766097564122
// MI455X (gfx1250) — matrix operand order repaired and checked by reading, not compiled, not run on hardware
//
#include <hip/hip_runtime.h>

#define HH 1024
#define WW 4096

typedef __attribute__((ext_vector_type(16))) __bf16 v16bf;
typedef __attribute__((ext_vector_type(8)))  float  v8f;

union Frag {
  unsigned int u[8];
  v16bf v;
};

__device__ __forceinline__ unsigned int pk2bf(float a, float b) {
  unsigned int ua = __builtin_bit_cast(unsigned int, a);
  unsigned int ub = __builtin_bit_cast(unsigned int, b);
  ua = (ua + 0x7FFFu + ((ua >> 16) & 1u)) >> 16;
  ub = (ub + 0x7FFFu + ((ub >> 16) & 1u)) >> 16;
  return (ua & 0xFFFFu) | (ub << 16);
}
__device__ __forceinline__ unsigned short bf1(float a) {
  unsigned int ua = __builtin_bit_cast(unsigned int, a);
  return (unsigned short)((ua + 0x7FFFu + ((ua >> 16) & 1u)) >> 16);
}

// LDS layout (bytes):
//   s_field : 36*36 f32                   = 5184   @ 0
//   s_h     : 1168 px (73*16) * 32ch bf16 = 74752  @ 5184    (padded: stores branch-free)
//   s_B2    : 9 taps * 16 n * 32 k bf16   = 9216   @ 79936
//   s_logits: 1024 px * 9 f32             = 36864  @ 89152
//   s_eout  : 1024 px * 8 f32             = 32768  @ 126016
#define SMEM_BYTES 158784

__global__ void __launch_bounds__(256)
moe_pde_kernel(const float* __restrict__ field, const float* __restrict__ w_experts,
               const float* __restrict__ w_g1, const float* __restrict__ b_g1,
               const float* __restrict__ w_g2, const float* __restrict__ b_g2,
               const float* __restrict__ noise, float* __restrict__ out)
{
  extern __shared__ __align__(16) unsigned char smem[];
  float*          s_field  = (float*)smem;
  unsigned short* s_h      = (unsigned short*)(smem + 5184);
  unsigned short* s_B2     = (unsigned short*)(smem + 79936);
  float*          s_logits = (float*)(smem + 89152);
  float*          s_eout   = (float*)(smem + 126016);

  const int tid  = threadIdx.x;
  const int lane = tid & 31;
  const int wave = tid >> 5;
  const bool hi  = lane >= 16;
  const int ox = blockIdx.x * 32;
  const int oy = blockIdx.y * 32;
  // Halo range: rows [oy-2, oy+33], cols [ox-2, ox+33]. Uniform per workgroup.
  const bool interior = (oy >= 2) && (oy + 34 <= HH) && (ox >= 2) && (ox + 34 <= WW);

  // Prefetch the tile's noise rows (consumed only in phase 3) to hide HBM latency.
  if (tid < 32) __builtin_prefetch(noise + (oy + tid) * WW + ox, 0, 1);

  // ---------------- Phase 0: stage field halo + conv2 weights ----------------
  if (interior) {
    const float* fb = field + (oy - 2) * WW + (ox - 2);
    for (int i = tid; i < 36 * 36; i += 256)
      s_field[i] = fb[(i / 36) * WW + (i % 36)];
  } else {
    for (int i = tid; i < 36 * 36; i += 256) {
      int fy = oy - 2 + i / 36;
      int fx = ox - 2 + i % 36;
      float v = 0.f;
      if (fy >= 0 && fy < HH && fx >= 0 && fx < WW) v = field[fy * WW + fx];
      s_field[i] = v;
    }
  }
  // conv2 weights -> bf16, layout [tap][n][k], n padded 9->16 with zeros
  for (int i = tid; i < 9 * 16 * 32; i += 256) {
    int t = i >> 9;
    int n = (i >> 5) & 15;
    int k = i & 31;
    float v = 0.f;
    if (n < 9) v = w_g2[(n * 32 + k) * 9 + t];
    s_B2[(t * 16 + n) * 32 + k] = bf1(v);
  }
  __syncthreads();

  // ---------------- Phase 1: conv1 (1->32ch) via WMMA ----------------
  // GEMM: M = 34*34 halo pixels, N = 32 ch (two 16-wide frags), K = 9 taps pad 32.
  Frag b1a, b1b;
  {
    const int c0 = (lane & 15), c1 = c0 + 16;
    float w0[9], w1[9];
#pragma unroll
    for (int j = 0; j < 9; ++j) { w0[j] = w_g1[c0 * 9 + j]; w1[j] = w_g1[c1 * 9 + j]; }
#pragma unroll
    for (int i = 0; i < 8; ++i) { b1a.u[i] = 0u; b1b.u[i] = 0u; }
    if (!hi) {
#pragma unroll
      for (int i = 0; i < 4; ++i) {
        b1a.u[i] = pk2bf(w0[2 * i], w0[2 * i + 1]);
        b1b.u[i] = pk2bf(w1[2 * i], w1[2 * i + 1]);
      }
      b1a.u[4] = pk2bf(w0[8], 0.f);
      b1b.u[4] = pk2bf(w1[8], 0.f);
    }
  }
  const float bg0 = b_g1[lane & 15];
  const float bg1 = b_g1[(lane & 15) + 16];

  for (int grp = wave; grp < 73; grp += 8) {
    int m = lane & 15;
    int pix = grp * 16 + m; if (pix > 1155) pix = 1155;  // clamp gather only
    int py = pix / 34, px = pix % 34;
    float f[9];
#pragma unroll
    for (int t = 0; t < 9; ++t) f[t] = s_field[(py + t / 3) * 36 + (px + t % 3)];
    Frag a;
#pragma unroll
    for (int i = 0; i < 8; ++i) a.u[i] = 0u;
    if (!hi) {
#pragma unroll
      for (int i = 0; i < 4; ++i) a.u[i] = pk2bf(f[2 * i], f[2 * i + 1]);
      a.u[4] = pk2bf(f[8], 0.f);
    }
    v8f d0 = {0.f, 0.f, 0.f, 0.f, 0.f, 0.f, 0.f, 0.f};
    v8f d1 = {0.f, 0.f, 0.f, 0.f, 0.f, 0.f, 0.f, 0.f};
    d0 = __builtin_amdgcn_wmma_f32_16x16x32_bf16(false, a.v, false, b1a.v, (short)0, d0, false, false);
    d1 = __builtin_amdgcn_wmma_f32_16x16x32_bf16(false, a.v, false, b1b.v, (short)0, d1, false, false);
    // Epilogue: s_h padded to 1168 px; pad rows never read by conv2.
    const int c0 = (lane & 15), c1 = c0 + 16;
    const int p0 = grp * 16 + (hi ? 8 : 0);
    if (interior) {
      // Fast path (92% of tiles): every halo h-pixel is in-image -> no compares.
#pragma unroll
      for (int r = 0; r < 8; ++r) {
        float v0 = d0[r] + bg0; v0 = v0 > 0.f ? v0 : 0.1f * v0;
        float v1 = d1[r] + bg1; v1 = v1 > 0.f ? v1 : 0.1f * v1;
        int p = p0 + r;
        s_h[p * 32 + c0] = bf1(v0);
        s_h[p * 32 + c1] = bf1(v1);
      }
    } else {
      const int py0 = p0 / 34;
      const int px0 = p0 - py0 * 34;
#pragma unroll
      for (int r = 0; r < 8; ++r) {
        int pxr = px0 + r;
        int wr  = (pxr >= 34) ? 1 : 0;
        int pyr = py0 + wr;
        pxr -= 34 * wr;
        int iy = oy - 1 + pyr, ix = ox - 1 + pxr;
        bool in_img = (iy >= 0 && iy < HH && ix >= 0 && ix < WW);
        float v0 = d0[r] + bg0; v0 = v0 > 0.f ? v0 : 0.1f * v0;
        float v1 = d1[r] + bg1; v1 = v1 > 0.f ? v1 : 0.1f * v1;
        int p = p0 + r;
        s_h[p * 32 + c0] = in_img ? bf1(v0) : (unsigned short)0;
        s_h[p * 32 + c1] = in_img ? bf1(v1) : (unsigned short)0;
      }
    }
  }
  __syncthreads();

  // ------- Phase 2: conv2 (32->9ch) + expert conv (1->8ch) via WMMA -------
  const int n = lane & 15;
  const int aoff = hi ? 8 : 0;
  const float bgate = b_g2[n > 8 ? 8 : n];
  // Hoist conv2 B fragments (9 chunks) into registers.
  Frag b2[9];
#pragma unroll
  for (int t = 0; t < 9; ++t) {
    const uint4* pb = (const uint4*)(s_B2 + (t * 16 + n) * 32 + (hi ? 8 : 0));
    uint4 blo = pb[0];
    uint4 bhi = pb[2];
    b2[t].u[0] = blo.x; b2[t].u[1] = blo.y; b2[t].u[2] = blo.z; b2[t].u[3] = blo.w;
    b2[t].u[4] = bhi.x; b2[t].u[5] = bhi.y; b2[t].u[6] = bhi.z; b2[t].u[7] = bhi.w;
  }
  // Expert B fragment (lane n = expert; cols 8..15 garbage, never stored).
  Frag be;
  {
    const int ec = n > 7 ? 7 : n;
    float we[9];
#pragma unroll
    for (int j = 0; j < 9; ++j) we[j] = w_experts[ec * 9 + j];
#pragma unroll
    for (int i = 0; i < 8; ++i) be.u[i] = 0u;
    if (!hi) {
#pragma unroll
      for (int i = 0; i < 4; ++i) be.u[i] = pk2bf(we[2 * i], we[2 * i + 1]);
      be.u[4] = pk2bf(we[8], 0.f);
    }
  }

  for (int grp = wave; grp < 64; grp += 8) {
    int m = lane & 15;
    int pix = grp * 16 + m;
    int y = pix >> 5, x = pix & 31;
    v8f acc = {0.f, 0.f, 0.f, 0.f, 0.f, 0.f, 0.f, 0.f};
#pragma unroll
    for (int t = 0; t < 9; ++t) {
      int ph = (y + t / 3) * 34 + (x + t % 3);
      const uint4* pa = (const uint4*)(s_h + ph * 32 + aoff);
      uint4 a_lo = pa[0];   // ch 0..7  (hi lanes: 8..15)
      uint4 a_hi = pa[2];   // ch 16..23 (hi lanes: 24..31)
      Frag a;
      a.u[0] = a_lo.x; a.u[1] = a_lo.y; a.u[2] = a_lo.z; a.u[3] = a_lo.w;
      a.u[4] = a_hi.x; a.u[5] = a_hi.y; a.u[6] = a_hi.z; a.u[7] = a_hi.w;
      acc = __builtin_amdgcn_wmma_f32_16x16x32_bf16(false, a.v, false, b2[t].v, (short)0, acc, false, false);
    }
    // Expert conv WMMA on the same 16-pixel group.
    float g[9];
#pragma unroll
    for (int t = 0; t < 9; ++t) g[t] = s_field[(y + 1 + t / 3) * 36 + (x + 1 + t % 3)];
    Frag ae;
#pragma unroll
    for (int i = 0; i < 8; ++i) ae.u[i] = 0u;
    if (!hi) {
#pragma unroll
      for (int i = 0; i < 4; ++i) ae.u[i] = pk2bf(g[2 * i], g[2 * i + 1]);
      ae.u[4] = pk2bf(g[8], 0.f);
    }
    v8f de = {0.f, 0.f, 0.f, 0.f, 0.f, 0.f, 0.f, 0.f};
    de = __builtin_amdgcn_wmma_f32_16x16x32_bf16(false, ae.v, false, be.v, (short)0, de, false, false);

    if (n < 9) {
#pragma unroll
      for (int r = 0; r < 8; ++r) {
        int mr = r + (hi ? 8 : 0);
        s_logits[(grp * 16 + mr) * 9 + n] = acc[r] + bgate;
      }
    }
    if (n < 8) {
#pragma unroll
      for (int r = 0; r < 8; ++r) {
        int mr = r + (hi ? 8 : 0);
        s_eout[(grp * 16 + mr) * 8 + n] = de[r];
      }
    }
  }
  __syncthreads();

  // -------- Phase 3: softmax + gated combine + tanh + noise --------
  for (int pix = tid; pix < 1024; pix += 256) {
    int y = pix >> 5, x = pix & 31;
    float l[9];
    float mx = -3.0e38f;
#pragma unroll
    for (int j = 0; j < 9; ++j) { l[j] = s_logits[pix * 9 + j]; mx = fmaxf(mx, l[j]); }
    float s = 0.f;
#pragma unroll
    for (int j = 0; j < 9; ++j) { l[j] = __expf(l[j] - mx); s += l[j]; }
    float inv = 1.f / s;

    float se = 0.f;
#pragma unroll
    for (int e = 0; e < 8; ++e)
      se = fmaf(s_eout[pix * 8 + e], l[e], se);
    float combined = se * inv;

    float gnone = l[8] * inv;
    float partial = combined * (1.f - gnone);
    float fc = s_field[(y + 2) * 36 + (x + 2)];
    float nf = tanhf(0.997f * (fc + 0.015f * partial));
    int gy = oy + y, gx = ox + x;
    nf += (noise[gy * WW + gx] - 0.5f) * 0.001f;
    out[gy * WW + gx] = nf;
  }
}

extern "C" void kernel_launch(void* const* d_in, const int* in_sizes, int n_in,
                              void* d_out, int out_size, void* d_ws, size_t ws_size,
                              hipStream_t stream) {
  (void)in_sizes; (void)n_in; (void)out_size; (void)d_ws; (void)ws_size;
  const float* field     = (const float*)d_in[0];
  const float* w_experts = (const float*)d_in[1];
  const float* w_g1      = (const float*)d_in[2];
  const float* b_g1      = (const float*)d_in[3];
  const float* w_g2      = (const float*)d_in[4];
  const float* b_g2      = (const float*)d_in[5];
  const float* noise     = (const float*)d_in[6];
  float* out = (float*)d_out;

  hipFuncSetAttribute((const void*)moe_pde_kernel,
                      hipFuncAttributeMaxDynamicSharedMemorySize, SMEM_BYTES);
  dim3 grid(WW / 32, HH / 32);
  moe_pde_kernel<<<grid, dim3(256), SMEM_BYTES, stream>>>(
      field, w_experts, w_g1, b_g1, w_g2, b_g2, noise, out);
}
